// SparseHashedNNDistance_52613349376207
// MI455X (gfx1250) — compile-verified, matrix operand order checked by reading, not run on hardware
//
#include <hip/hip_runtime.h>
#include <hip/hip_bf16.h>
#include <math.h>

#define BDIM  8
#define NPTS  10000
#define NF    16
#define DDIM  128
#define NBINS 20
#define BINSZ 500
#define TOPK  5
#define RCOLS 100            // R is (128, 100); we use first 10 cols
#define EPB   (NPTS * TOPK)  // edges per batch = 50000

typedef __attribute__((ext_vector_type(16))) __bf16 v16bf;
typedef __attribute__((ext_vector_type(8)))  float  v8f;

union Frag { v16bf v; unsigned u[8]; unsigned short h[16]; };

__device__ __forceinline__ unsigned short f2bf(float f) {
  union { float f; unsigned u; } x; x.f = f;
  unsigned u = x.u;
  u += 0x7FFFu + ((u >> 16) & 1u);  // round-to-nearest-even
  return (unsigned short)(u >> 16);
}

// Per-wave LDS fence: CDNA5 LDS ops from one wave are processed in order
// (ISA 7.3), so a compiler scheduling barrier is sufficient for intra-wave
// store->load visibility through per-wave LDS tiles. Zero instructions.
__device__ __forceinline__ void wave_lds_fence() {
  __builtin_amdgcn_wave_barrier();
}

// 16-bit A/B fragment K-offset for VGPR v on this lane (CDNA5 16x32 layout):
// lanes 0-15 hold K-halves {0..7,16..23}, lanes 16-31 hold {8..15,24..31}.
__device__ __forceinline__ int frag_k0(int v, int lane) {
  int khalf = (lane & 16) ? 8 : 0;
  int base  = (v < 4) ? (2 * v) : (16 + 2 * (v - 4));
  return base + khalf;
}

// Load one 16x32 bf16 fragment from row-major bf16 storage (ld elements,
// ld and kbase even). Works for A (row = M) and B (row = N of transposed W).
// The two 4-dword groups coalesce into two b128 loads (seen in codegen).
__device__ __forceinline__ Frag load_frag(const unsigned short* p, int ld,
                                          int row, int kbase, int lane) {
  Frag f;
  const unsigned* pr = (const unsigned*)(p + (size_t)row * ld + kbase);
#pragma unroll
  for (int v = 0; v < 8; ++v) f.u[v] = pr[frag_k0(v, lane) >> 1];
  return f;
}

__device__ __forceinline__ Frag zero_frag() {
  Frag f;
#pragma unroll
  for (int v = 0; v < 8; ++v) f.u[v] = 0u;
  return f;
}

// ---------------------------------------------------------------- prep ------
__global__ void prep_kernel(const float* W1, const float* W2, const float* We1,
                            unsigned short* W1t, unsigned short* W2t,
                            unsigned short* We1t, int* hist) {
  int t = threadIdx.x;
  for (int i = t; i < 128 * 32; i += blockDim.x) {          // W_enc1^T, K pad 16->32
    int n = i >> 5, k = i & 31;
    W1t[i] = f2bf(k < NF ? W1[k * DDIM + n] : 0.f);
  }
  for (int i = t; i < 128 * 128; i += blockDim.x) {         // W_enc2^T
    int n = i >> 7, k = i & 127;
    W2t[i] = f2bf(W2[k * DDIM + n]);
  }
  for (int i = t; i < 128 * 64; i += blockDim.x) {          // W_edge1^T, K pad 33->64
    int n = i >> 6, k = i & 63;
    We1t[i] = f2bf(k < 33 ? We1[k * DDIM + n] : 0.f);
  }
  for (int i = t; i < BDIM * NBINS; i += blockDim.x) hist[i] = 0;
}

// ------------------------------------------------------------- encoder ------
// One wave per 16-point tile: X(16x16 pad32) @ W1(32x128) -> ELU -> @ W2(128x128)
__global__ void __launch_bounds__(256)
enc_kernel(const float* x, const unsigned short* W1t, const float* b1,
           const unsigned short* W2t, const float* b2,
           float* encF, unsigned short* encH) {
  __shared__ __align__(16) unsigned short ldsX[8][16 * 32];
  __shared__ __align__(16) unsigned short ldsH[8][16 * 128];
  int lane = threadIdx.x & 31;
  int w    = threadIdx.x >> 5;
  int pbase = (blockIdx.x * 8 + w) * 16;
  int n0 = lane & 15;

  for (int i = lane; i < 16 * 32; i += 32) {        // stage X as bf16, K-padded
    int r = i >> 5, c = i & 31;
    ldsX[w][i] = f2bf(c < NF ? x[(size_t)(pbase + r) * NF + c] : 0.f);
  }
  wave_lds_fence();

  Frag a1 = load_frag(ldsX[w], 32, n0, 0, lane);

#pragma unroll
  for (int t = 0; t < 8; ++t) {                     // layer 1 + bias + ELU
    Frag b = load_frag(W1t, 32, t * 16 + n0, 0, lane);
    v8f c = {};
    c = __builtin_amdgcn_wmma_f32_16x16x32_bf16(false, a1.v, false, b.v,
                                                (short)0, c, false, false);
    int col = t * 16 + n0;
    float bias = b1[col];
#pragma unroll
    for (int j = 0; j < 8; ++j) {
      int m = j + ((lane & 16) ? 8 : 0);
      float h = c[j] + bias;
      h = h > 0.f ? h : (__expf(h) - 1.f);
      ldsH[w][m * 128 + col] = f2bf(h);
    }
  }
  wave_lds_fence();

  Frag a2[4];
#pragma unroll
  for (int kc = 0; kc < 4; ++kc) a2[kc] = load_frag(ldsH[w], 128, n0, kc * 32, lane);

#pragma unroll
  for (int t = 0; t < 8; ++t) {                     // layer 2
    v8f c = {};
#pragma unroll
    for (int kc = 0; kc < 4; ++kc) {
      Frag b = load_frag(W2t, 128, t * 16 + n0, kc * 32, lane);
      c = __builtin_amdgcn_wmma_f32_16x16x32_bf16(false, a2[kc].v, false, b.v,
                                                  (short)0, c, false, false);
    }
    int col = t * 16 + n0;
    float bias = b2[col];
#pragma unroll
    for (int j = 0; j < 8; ++j) {
      int m = j + ((lane & 16) ? 8 : 0);
      float v = c[j] + bias;
      size_t p = (size_t)(pbase + m) * DDIM + col;
      encF[p] = v;
      encH[p] = f2bf(v);
    }
  }
}

// ------------------------------------------------------- binning + hist -----
__global__ void bin_kernel(const float* encF, const float* R, int* binIdx, int* hist) {
  int p = blockIdx.x * blockDim.x + threadIdx.x;
  if (p >= BDIM * NPTS) return;
  const float* e = encF + (size_t)p * DDIM;
  float mul[10];
#pragma unroll
  for (int j = 0; j < 10; ++j) mul[j] = 0.f;
  for (int d = 0; d < DDIM; ++d) {
    float ed = e[d];
    const float* r = R + (size_t)d * RCOLS;
#pragma unroll
    for (int j = 0; j < 10; ++j) mul[j] += ed * r[j];
  }
  // argmax over concat([mul, -mul]), first-max wins (strict >)
  float best = mul[0]; int bi = 0;
  for (int j = 1; j < 10; ++j) if (mul[j] > best) { best = mul[j]; bi = j; }
  for (int j = 0; j < 10; ++j) if (-mul[j] > best) { best = -mul[j]; bi = 10 + j; }
  binIdx[p] = bi;
  atomicAdd(&hist[(p / NPTS) * NBINS + bi], 1);
}

__global__ void scan_kernel(const int* hist, int* offs) {
  int b = threadIdx.x;
  if (b >= BDIM) return;
  int s = 0;
  for (int j = 0; j < NBINS; ++j) { offs[b * NBINS + j] = s; s += hist[b * NBINS + j]; }
}

// Stable counting-sort scatter: one wave per (batch, bin); ballot keeps
// index order, so concatenation == stable argsort(bin_idx).
__global__ void scatter_kernel(const int* binIdx, const int* offs, int* binsSplit) {
  int b = blockIdx.x / NBINS, bin = blockIdx.x % NBINS;
  int lane = threadIdx.x;
  int base = offs[b * NBINS + bin];
  const int* bi = binIdx + (size_t)b * NPTS;
  for (int i = 0; i < NPTS; i += 32) {
    int p = i + lane;
    bool pred = (p < NPTS) && (bi[p] == bin);
    unsigned long long mask = __ballot(pred);
    if (pred) {
      int rank = __popcll(mask & ((1ull << lane) - 1ull));
      binsSplit[(size_t)b * NPTS + base + rank] = p;
    }
    base += __popcll(mask);
  }
}

// ------------------------------------------- per-chunk similarity + top-5 ---
// Block per (batch, chunk); wave per 16-row tile (4 tiles/wave). 512x512 padded
// GEMM via bf16 WMMA (K=128 -> 4 wmma per 16x16 tile). Top-5 runs on RAW dot
// products (sigmoid is monotonic -> applied to only 5 values/row at the end).
// All 32 lanes merge: lane owns (row = lane&15, col-half = lane>>4).
__global__ void __launch_bounds__(256)
sim_kernel(const unsigned short* encH, const int* binsSplit, float* val5, int* dst5) {
  __shared__ __align__(16) float ldsS[8][256];
  int b = blockIdx.x / NBINS, bin = blockIdx.x % NBINS;
  int lane = threadIdx.x & 31;
  int w    = threadIdx.x >> 5;
  int m    = lane & 15;
  int h8   = (lane & 16) ? 8 : 0;
  const int* split = binsSplit + (size_t)b * NPTS + bin * BINSZ;
  const unsigned short* eb = encH + (size_t)b * NPTS * DDIM;

  for (int rt = w; rt < 32; rt += 8) {
    int rbase = rt * 16;
    int rloc  = rbase + m;
    int rowg  = (rloc < BINSZ) ? split[rloc] : -1;
    Frag a[4];
#pragma unroll
    for (int kc = 0; kc < 4; ++kc)
      a[kc] = (rowg >= 0) ? load_frag(eb, DDIM, rowg, kc * 32, lane) : zero_frag();

    float tv[TOPK]; int tc[TOPK];
#pragma unroll
    for (int k = 0; k < TOPK; ++k) { tv[k] = -1e30f; tc[k] = 0; }

    for (int ct = 0; ct < 32; ++ct) {
      int cbase = ct * 16;
      int cloc  = cbase + m;
      int colg  = (cloc < BINSZ) ? split[cloc] : -1;

      // prefetch next column tile's rows (chunk streams from L2 each pass)
      if (ct + 1 < 32) {
        int nloc = cbase + 16 + m;
        if (nloc < BINSZ) {
          const unsigned short* nr = eb + (size_t)split[nloc] * DDIM;
          __builtin_prefetch(nr, 0, 3);
          __builtin_prefetch(nr + 64, 0, 3);
        }
      }

      v8f c = {};
#pragma unroll
      for (int kc = 0; kc < 4; ++kc) {
        Frag bf = (colg >= 0) ? load_frag(eb, DDIM, colg, kc * 32, lane) : zero_frag();
        c = __builtin_amdgcn_wmma_f32_16x16x32_bf16(false, a[kc].v, false, bf.v,
                                                    (short)0, c, false, false);
      }
#pragma unroll
      for (int j = 0; j < 8; ++j) {                 // raw scores, D-frag -> LDS tile
        int mm = j + h8;
        ldsS[w][mm * 16 + m] = c[j];
      }
      wave_lds_fence();
      {                                             // each lane scans 8 cols of its row
        const float* row = &ldsS[w][m * 16 + h8];
#pragma unroll
        for (int n = 0; n < 8; ++n) {
          int col = cbase + h8 + n;
          if (col < BINSZ) {
            float val = row[n];
            if (val > tv[TOPK - 1]) {               // strict >: lower index wins ties
              int pos = TOPK - 1;
              while (pos > 0 && val > tv[pos - 1]) {
                tv[pos] = tv[pos - 1]; tc[pos] = tc[pos - 1]; --pos;
              }
              tv[pos] = val; tc[pos] = col;
            }
          }
        }
      }
      wave_lds_fence();                             // before next tile overwrites
    }

    // merge the two column-half top-5 lists (tie-break: lower column index)
    float ov[TOPK]; int oc[TOPK];
#pragma unroll
    for (int k = 0; k < TOPK; ++k) {
      ov[k] = __shfl_xor(tv[k], 16, 32);
      oc[k] = __shfl_xor(tc[k], 16, 32);
    }
    if (lane < 16 && rbase + lane < BINSZ) {
      float mv[TOPK]; int mc[TOPK];
      int i = 0, j = 0;
#pragma unroll
      for (int k = 0; k < TOPK; ++k) {
        bool takeA = (j >= TOPK) ||
                     ((i < TOPK) && ((tv[i] > ov[j]) ||
                                     (tv[i] == ov[j] && tc[i] < oc[j])));
        if (takeA) { mv[k] = tv[i]; mc[k] = tc[i]; ++i; }
        else       { mv[k] = ov[j]; mc[k] = oc[j]; ++j; }
      }
      int gsrc = split[rbase + lane];
      int gd[TOPK]; float gv[TOPK];
#pragma unroll
      for (int k = 0; k < TOPK; ++k) {
        gd[k] = split[mc[k]];
        gv[k] = 1.f / (1.f + __expf(-mv[k]));       // sigmoid only on the 5 winners
      }
      for (int i2 = 1; i2 < TOPK; ++i2) {           // sort 5 by dst ascending
        int dk = gd[i2]; float vk = gv[i2]; int j2 = i2 - 1;
        while (j2 >= 0 && gd[j2] > dk) { gd[j2 + 1] = gd[j2]; gv[j2 + 1] = gv[j2]; --j2; }
        gd[j2 + 1] = dk; gv[j2 + 1] = vk;
      }
      size_t o = ((size_t)b * NPTS + gsrc) * TOPK;  // == global argsort position
#pragma unroll
      for (int k = 0; k < TOPK; ++k) { val5[o + k] = gv[k]; dst5[o + k] = gd[k]; }
    }
  }
}

// ------------------------------------------------------------ edge MLP ------
// One wave per 16-edge tile: in(16x33 pad64) @ We1 -> ELU -> dot We2 -> sigmoid
__global__ void __launch_bounds__(256)
edge_kernel(const float* x, const unsigned short* We1t, const float* be1,
            const float* We2, const float* be2,
            const float* val5, const int* dst5,
            float* edgeOut, int* srcOut, int* dstOut) {
  __shared__ __align__(16) unsigned short ldsE[8][16 * 64];
  int lane = threadIdx.x & 31;
  int w    = threadIdx.x >> 5;
  int ebase = (blockIdx.x * 8 + w) * 16;
  int n0 = lane & 15;

  for (int i = lane; i < 16 * 64; i += 32) {        // build [x[src], x[dst], val, 0pad]
    int r = i >> 6, c = i & 63;
    int e = ebase + r;
    int b = e / EPB, rr = e - b * EPB;
    int src = rr / TOPK;
    float v;
    if (c < NF)            v = x[((size_t)b * NPTS + src) * NF + c];
    else if (c < 2 * NF)   v = x[((size_t)b * NPTS + dst5[e]) * NF + (c - NF)];
    else if (c == 2 * NF)  v = val5[e];
    else                   v = 0.f;
    ldsE[w][i] = f2bf(v);
  }
  wave_lds_fence();

  Frag a0 = load_frag(ldsE[w], 64, n0, 0, lane);
  Frag a1 = load_frag(ldsE[w], 64, n0, 32, lane);

  float partial[8];
#pragma unroll
  for (int j = 0; j < 8; ++j) partial[j] = 0.f;

#pragma unroll
  for (int t = 0; t < 8; ++t) {
    v8f c = {};
    Frag bf0 = load_frag(We1t, 64, t * 16 + n0, 0, lane);
    c = __builtin_amdgcn_wmma_f32_16x16x32_bf16(false, a0.v, false, bf0.v,
                                                (short)0, c, false, false);
    Frag bf1 = load_frag(We1t, 64, t * 16 + n0, 32, lane);
    c = __builtin_amdgcn_wmma_f32_16x16x32_bf16(false, a1.v, false, bf1.v,
                                                (short)0, c, false, false);
    int col = t * 16 + n0;
    float bias = be1[col];
    float w2 = We2[col];
#pragma unroll
    for (int j = 0; j < 8; ++j) {                   // fused ELU + layer-2 dot
      float h = c[j] + bias;
      h = h > 0.f ? h : (__expf(h) - 1.f);
      partial[j] += h * w2;
    }
  }
#pragma unroll
  for (int j = 0; j < 8; ++j)                       // reduce across 16-lane halves
#pragma unroll
    for (int msk = 1; msk < 16; msk <<= 1)
      partial[j] += __shfl_xor(partial[j], msk, 32);

  float bias2 = be2[0];
  if (lane == 0 || lane == 16) {
    int mbase = (lane == 16) ? 8 : 0;
#pragma unroll
    for (int j = 0; j < 8; ++j)
      edgeOut[ebase + mbase + j] = 1.f / (1.f + __expf(-(partial[j] + bias2)));
  }
  if (lane < 16) {
    int e = ebase + lane;
    int b = e / EPB, rr = e - b * EPB;
    srcOut[e] = rr / TOPK;
    dstOut[e] = dst5[e];
  }
}

// --------------------------------------------------------------- launch -----
extern "C" void kernel_launch(void* const* d_in, const int* in_sizes, int n_in,
                              void* d_out, int out_size, void* d_ws, size_t ws_size,
                              hipStream_t stream) {
  (void)in_sizes; (void)n_in; (void)out_size; (void)ws_size;
  const float* x   = (const float*)d_in[0];
  const float* W1  = (const float*)d_in[1];
  const float* b1  = (const float*)d_in[2];
  const float* W2  = (const float*)d_in[3];
  const float* b2  = (const float*)d_in[4];
  const float* We1 = (const float*)d_in[5];
  const float* be1 = (const float*)d_in[6];
  const float* We2 = (const float*)d_in[7];
  const float* be2 = (const float*)d_in[8];
  const float* R   = (const float*)d_in[9];

  char* ws = (char*)d_ws;
  size_t off = 0;
  auto alloc = [&](size_t bytes) -> void* {
    void* p = ws + off;
    off += (bytes + 255) & ~(size_t)255;
    return p;
  };
  float*          encF      = (float*)          alloc(sizeof(float) * BDIM * NPTS * DDIM);
  unsigned short* encH      = (unsigned short*) alloc(2u * BDIM * NPTS * DDIM);
  unsigned short* W1t       = (unsigned short*) alloc(2u * 128 * 32);
  unsigned short* W2t       = (unsigned short*) alloc(2u * 128 * 128);
  unsigned short* We1t      = (unsigned short*) alloc(2u * 128 * 64);
  int*            binIdx    = (int*)            alloc(sizeof(int) * BDIM * NPTS);
  int*            hist      = (int*)            alloc(sizeof(int) * BDIM * NBINS);
  int*            offs      = (int*)            alloc(sizeof(int) * BDIM * NBINS);
  int*            binsSplit = (int*)            alloc(sizeof(int) * BDIM * NPTS);
  float*          val5      = (float*)          alloc(sizeof(float) * BDIM * NPTS * TOPK);
  int*            dst5      = (int*)            alloc(sizeof(int) * BDIM * NPTS * TOPK);

  float* edgeOut = (float*)d_out;
  int*   srcOut  = (int*)d_out + (size_t)BDIM * EPB;
  int*   dstOut  = (int*)d_out + 2u * (size_t)BDIM * EPB;

  prep_kernel<<<1, 256, 0, stream>>>(W1, W2, We1, W1t, W2t, We1t, hist);
  enc_kernel<<<(BDIM * NPTS / 16) / 8, 256, 0, stream>>>(x, W1t, b1, W2t, b2, encF, encH);
  bin_kernel<<<(BDIM * NPTS + 255) / 256, 256, 0, stream>>>(encF, R, binIdx, hist);
  scan_kernel<<<1, 32, 0, stream>>>(hist, offs);
  scatter_kernel<<<BDIM * NBINS, 32, 0, stream>>>(binIdx, offs, binsSplit);
  sim_kernel<<<BDIM * NBINS, 256, 0, stream>>>(encH, binsSplit, val5, dst5);
  edge_kernel<<<(BDIM * EPB / 16) / 8, 256, 0, stream>>>(x, We1t, be1, We2, be2,
                                                         val5, dst5, edgeOut, srcOut, dstOut);
}
